// FuXiLowerResDecoder_79577154060526
// MI455X (gfx1250) — compile-verified, matrix operand order checked by reading, not run on hardware
//
#include <hip/hip_runtime.h>
#include <hip/hip_bf16.h>

// ---------------- static config ----------------
#define H_LAT 27
#define W_LAT 54
#define EMBED 1536
#define HEADS 24
#define HEAD_DIM 64
#define WS9 9
#define NTOK 81
#define SHIFT4 4
#define T_TOK 2916      // B * 27 * 54
#define TUP 11664       // B * 54 * 108
#define BATCH 2
#define MLP_HID 4096
#define LOGIT_MAX 4.605170185988092f

#define USE_ASYNC_LDS 1

typedef _Float16 half16 __attribute__((ext_vector_type(16)));
typedef _Float16 half8  __attribute__((ext_vector_type(8)));
typedef float    f32x8  __attribute__((ext_vector_type(8)));
typedef __attribute__((address_space(3))) _Float16 lds_f16;

union Frag16 { half16 v; half8 h[2]; };

// ---------------- generic elementwise ----------------
__global__ __launch_bounds__(256)
void k_f32_to_f16(const float* __restrict__ s, _Float16* __restrict__ d, long n) {
  long i = (long)blockIdx.x * blockDim.x + threadIdx.x;
  if (i < n) d[i] = (_Float16)s[i];
}

__global__ __launch_bounds__(256)
void k_copy_f32(const float* __restrict__ s, float* __restrict__ d, long n) {
  long i = (long)blockIdx.x * blockDim.x + threadIdx.x;
  if (i < n) d[i] = s[i];
}

__global__ __launch_bounds__(256)
void k_pack_qkvb(const float* __restrict__ qb, const float* __restrict__ vb, float* __restrict__ dst) {
  int i = blockIdx.x * blockDim.x + threadIdx.x;
  if (i >= 3 * EMBED) return;
  dst[i] = (i < EMBED) ? qb[i] : ((i < 2 * EMBED) ? 0.0f : vb[i - 2 * EMBED]);
}

// ---------------- WMMA GEMM: C[M,N] (+=) A[M,K] * B^T, B(n,k) = Bp[n*ldbn + k]
// Block tile 128x128 (8 waves), wave tile 32(M) x 64(N): 2 A-frags x 4 B-frags
// -> 8 v_wmma_f32_16x16x32_f16 per K-step; A staged in LDS via async DMA.
#define BM 128
#define BN 128
#define KSTEP 32

__global__ __launch_bounds__(256)
void k_gemm_nt(const _Float16* __restrict__ A, int lda,
               const _Float16* __restrict__ Bp, long ldbn,
               const float* __restrict__ bias,
               float* __restrict__ C, int ldc,
               int M, int N, int K, int accumulate, int act) {
  __shared__ _Float16 aTile[BM * KSTEP];        // 8 KB
  const int tid  = threadIdx.x;
  const int wave = tid >> 5;
  const int lane = tid & 31;
  const int m_blk = blockIdx.x * BM;
  const int n_blk = blockIdx.y * BN;
  const int wm = wave >> 1;                     // 0..3
  const int wn = wave & 1;                      // 0..1
  const int m0 = m_blk + wm * 32;
  const int n0 = n_blk + wn * 64;
  const int l16 = lane & 15;
  const int lhi = lane >> 4;                    // 0/1

  // A staging: 128 rows x 32 halves; thread -> row = tid>>1, 16 halves at (tid&1)*16
  const int arow = tid >> 1;
  const int acol = (tid & 1) * 16;
  int gr = m_blk + arow;
  if (gr >= M) gr = M - 1;                      // clamp: pad rows feed discarded C rows
  const _Float16* agp = A + (long)gr * lda + acol;
#if USE_ASYNC_LDS
  const unsigned lds0 = (unsigned)(unsigned long long)(lds_f16*)(aTile + arow * KSTEP + acol);
#endif

  f32x8 acc[2][4] = {};

  for (int kk = 0; kk < K; kk += KSTEP) {
#if USE_ASYNC_LDS
    {
      unsigned long long g0 = (unsigned long long)(const void*)(agp + kk);
      asm volatile("global_load_async_to_lds_b128 %0, %1, off"
                   :: "v"(lds0), "v"(g0) : "memory");
      asm volatile("global_load_async_to_lds_b128 %0, %1, off offset:16"
                   :: "v"(lds0), "v"(g0) : "memory");
      asm volatile("s_wait_asynccnt 0" ::: "memory");
    }
#else
    {
      half8 av0 = *(const half8*)(agp + kk);
      half8 av1 = *(const half8*)(agp + kk + 8);
      *(half8*)(aTile + arow * KSTEP + acol) = av0;
      *(half8*)(aTile + arow * KSTEP + acol + 8) = av1;
    }
#endif
    __syncthreads();

    Frag16 a0, a1;
    {
      const _Float16* ap = aTile + (wm * 32 + l16) * KSTEP + lhi * 8;
      a0.h[0] = *(const half8*)(ap);
      a0.h[1] = *(const half8*)(ap + 16);
      const _Float16* ap1 = ap + 16 * KSTEP;
      a1.h[0] = *(const half8*)(ap1);
      a1.h[1] = *(const half8*)(ap1 + 16);
    }

#pragma unroll
    for (int f = 0; f < 4; ++f) {
      long col  = n0 + f * 16 + l16;
      long colc = (col < N) ? col : (N - 1);
      const _Float16* bp = Bp + colc * ldbn + kk + lhi * 8;
      Frag16 b;
      b.h[0] = *(const half8*)(bp);
      b.h[1] = *(const half8*)(bp + 16);
      __builtin_prefetch((const void*)(bp + KSTEP));    // global_prefetch_b8
      acc[0][f] = __builtin_amdgcn_wmma_f32_16x16x32_f16(false, a0.v, false, b.v,
                                                         (short)0, acc[0][f], false, false);
      acc[1][f] = __builtin_amdgcn_wmma_f32_16x16x32_f16(false, a1.v, false, b.v,
                                                         (short)0, acc[1][f], false, false);
    }
    __syncthreads();
  }

#pragma unroll
  for (int s = 0; s < 2; ++s) {
#pragma unroll
    for (int f = 0; f < 4; ++f) {
      int col = n0 + f * 16 + l16;
      if (col >= N) continue;
      float bv = bias ? bias[col] : 0.0f;
#pragma unroll
      for (int j = 0; j < 8; ++j) {
        int row = m0 + s * 16 + lhi * 8 + j;
        if (row >= M) continue;
        long idx = (long)row * ldc + col;
        float v = acc[s][f][j];
        if (accumulate) v += C[idx];
        v += bv;
        if (act == 1) v = v / (1.0f + expf(-v));   // SiLU
        C[idx] = v;
      }
    }
  }
}

// ---------------- token gathers / scatters ----------------
__global__ __launch_bounds__(256)
void k_gather_win(const float* __restrict__ x, _Float16* __restrict__ dst, int shift) {
  long i = (long)blockIdx.x * blockDim.x + threadIdx.x;
  long n = (long)T_TOK * EMBED;
  if (i >= n) return;
  long t = i / EMBED; int c = (int)(i % EMBED);
  int widx = (int)(t / NTOK), o = (int)(t % NTOK);
  int ii = o / WS9, jj = o % WS9;
  int b = widx / 18, wr = widx % 18, wh = wr / 6, ww = wr % 6;
  int h = (wh * WS9 + ii + shift) % H_LAT;
  int w = (ww * WS9 + jj + shift) % W_LAT;
  dst[i] = (_Float16)x[((long)(b * H_LAT + h) * W_LAT + w) * EMBED + c];
}

__global__ __launch_bounds__(256)
void k_gather_tap(const float* __restrict__ x, _Float16* __restrict__ dst,
                  int Bn, int C, int H, int W, int dy, int dx) {
  long n = (long)Bn * H * W * C;
  long i = (long)blockIdx.x * blockDim.x + threadIdx.x;
  if (i >= n) return;
  long t = i / C; int c = (int)(i % C);
  int b = (int)(t / (H * W)); int r = (int)(t % (H * W));
  int h = r / W + dy, w = r % W + dx;
  float v = 0.0f;
  if (h >= 0 && h < H && w >= 0 && w < W)
    v = x[((long)(b * C + c) * H + h) * W + w];
  dst[(long)t * C + c] = (_Float16)v;
}

__global__ __launch_bounds__(256)
void k_concat_tok_f16(const float* __restrict__ s2, const float* __restrict__ s3,
                      _Float16* __restrict__ dst) {
  long n = (long)T_TOK * 2 * EMBED;
  long i = (long)blockIdx.x * blockDim.x + threadIdx.x;
  if (i >= n) return;
  long t = i / (2 * EMBED); int c = (int)(i % (2 * EMBED));
  float v = (c < EMBED) ? s2[t * EMBED + c] : s3[t * EMBED + (c - EMBED)];
  dst[i] = (_Float16)v;
}

__global__ __launch_bounds__(256)
void k_concat_nchw(const float* __restrict__ tok, const float* __restrict__ skip,
                   float* __restrict__ dst) {
  long n = (long)BATCH * 2 * EMBED * H_LAT * W_LAT;
  long i = (long)blockIdx.x * blockDim.x + threadIdx.x;
  if (i >= n) return;
  int HW = H_LAT * W_LAT;
  int p = (int)(i % HW); long r = i / HW;
  int c = (int)(r % (2 * EMBED)); int b = (int)(r / (2 * EMBED));
  float v;
  if (c < EMBED) v = tok[((long)b * HW + p) * EMBED + c];
  else           v = skip[((long)b * EMBED + (c - EMBED)) * HW + p];
  dst[i] = v;
}

__global__ __launch_bounds__(256)
void k_tok_to_nchw(const float* __restrict__ tok, float* __restrict__ dst,
                   const float* __restrict__ add_tok, int C) {
  long n = (long)BATCH * C * H_LAT * W_LAT;
  long i = (long)blockIdx.x * blockDim.x + threadIdx.x;
  if (i >= n) return;
  int HW = H_LAT * W_LAT;
  int p = (int)(i % HW); long r = i / HW;
  int c = (int)(r % C); int b = (int)(r / C);
  long ti = ((long)b * HW + p) * C + c;
  float v = tok[ti];
  if (add_tok) v += add_tok[ti];
  dst[i] = v;
}

__global__ __launch_bounds__(256)
void k_scatterT(const float* __restrict__ yc, float* __restrict__ hup,
                int pa, int pb, const float* __restrict__ ub) {
  long n = (long)T_TOK * EMBED;
  long i = (long)blockIdx.x * blockDim.x + threadIdx.x;
  if (i >= n) return;
  long t = i / EMBED; int c = (int)(i % EMBED);
  int b = (int)(t / (H_LAT * W_LAT)); int r = (int)(t % (H_LAT * W_LAT));
  int q = r / W_LAT, s = r % W_LAT;
  int oh = 2 * q + pa, ow = 2 * s + pb;
  hup[(((long)b * EMBED + c) * 54 + oh) * 108 + ow] = yc[i] + ub[c];
}

__global__ __launch_bounds__(256)
void k_head_reorder(const float* __restrict__ ho, float* __restrict__ out, long n) {
  long i = (long)blockIdx.x * blockDim.x + threadIdx.x;
  if (i >= n) return;
  long tmp = i;
  int wpix = (int)(tmp % 432); tmp /= 432;
  int hpix = (int)(tmp % 216); tmp /= 216;
  int c = (int)(tmp % 70); tmp /= 70;
  int s = (int)(tmp % 2);  int b = (int)(tmp / 2);
  int oh = hpix >> 2, ph = hpix & 3, ow = wpix >> 2, pw = wpix & 3;
  out[i] = ho[((long)(b * 54 + oh) * 108 + ow) * 2240 + ((s * 70 + c) * 16 + ph * 4 + pw)];
}

// ---------------- norms ----------------
__global__ __launch_bounds__(256)
void k_add_ln(const float* __restrict__ src, const float* __restrict__ resid,
              float* __restrict__ dst, const float* __restrict__ gw,
              const float* __restrict__ gb, int win_mode, int shift) {
  int t = blockIdx.x;
  int tid = threadIdx.x;
  long srow;
  if (win_mode) {
    int b = t / (H_LAT * W_LAT); int r = t % (H_LAT * W_LAT);
    int h = r / W_LAT, w = r % W_LAT;
    int hh = (h - shift + H_LAT) % H_LAT;
    int wc = (w - shift + W_LAT) % W_LAT;
    srow = ((long)((b * 3 + hh / WS9) * 6 + wc / WS9)) * NTOK + (hh % WS9) * WS9 + (wc % WS9);
  } else srow = t;
  const float* sp = src + srow * EMBED;
  __shared__ float red[256];
  float s = 0.f;
  for (int c = tid; c < EMBED; c += 256) s += sp[c];
  red[tid] = s; __syncthreads();
  for (int o = 128; o > 0; o >>= 1) { if (tid < o) red[tid] += red[tid + o]; __syncthreads(); }
  float mu = red[0] / (float)EMBED;
  __syncthreads();
  float s2 = 0.f;
  for (int c = tid; c < EMBED; c += 256) { float d = sp[c] - mu; s2 += d * d; }
  red[tid] = s2; __syncthreads();
  for (int o = 128; o > 0; o >>= 1) { if (tid < o) red[tid] += red[tid + o]; __syncthreads(); }
  float rs = rsqrtf(red[0] / (float)EMBED + 1e-5f);
  for (int c = tid; c < EMBED; c += 256)
    dst[(long)t * EMBED + c] = resid[(long)t * EMBED + c] + (sp[c] - mu) * rs * gw[c] + gb[c];
}

__global__ __launch_bounds__(256)
void k_gn(const float* __restrict__ x, const float* __restrict__ gw, const float* __restrict__ gb,
          const float* __restrict__ ss, float* __restrict__ y, int C, int do_silu) {
  const int HW = H_LAT * W_LAT;
  int b = blockIdx.x >> 5, g = blockIdx.x & 31;
  int cpg = C >> 5;
  long base = ((long)b * C + (long)g * cpg) * HW;
  long n = (long)cpg * HW;
  int tid = threadIdx.x;
  float s = 0.f, s2 = 0.f;
  for (long i = tid; i < n; i += 256) { float v = x[base + i]; s += v; s2 += v * v; }
  __shared__ float r1[256], r2[256];
  r1[tid] = s; r2[tid] = s2; __syncthreads();
  for (int o = 128; o > 0; o >>= 1) {
    if (tid < o) { r1[tid] += r1[tid + o]; r2[tid] += r2[tid + o]; }
    __syncthreads();
  }
  float mu = r1[0] / (float)n;
  float var = r2[0] / (float)n - mu * mu;
  float rs = rsqrtf(var + 1e-5f);
  for (long i = tid; i < n; i += 256) {
    int c = g * cpg + (int)(i / HW);
    float v = (x[base + i] - mu) * rs * gw[c] + gb[c];
    if (ss) v = v * (1.f + ss[(long)b * 2 * C + c]) + ss[(long)b * 2 * C + C + c];
    if (do_silu) v = v / (1.f + expf(-v));
    y[base + i] = v;
  }
}

// ---------------- MLP pieces ----------------
__global__ __launch_bounds__(256)
void k_glu(const float* __restrict__ g, _Float16* __restrict__ out) {
  long n = (long)T_TOK * MLP_HID;
  long i = (long)blockIdx.x * blockDim.x + threadIdx.x;
  if (i >= n) return;
  long t = i / MLP_HID; int c = (int)(i % MLP_HID);
  float gate = g[t * 2 * MLP_HID + c];
  float val  = g[t * 2 * MLP_HID + MLP_HID + c];
  float ge = 0.5f * gate * (1.0f + erff(gate * 0.70710678118654752f));
  out[i] = (_Float16)(ge * val);
}

__global__ __launch_bounds__(256)
void k_ss(const float* __restrict__ temb, const float* __restrict__ w,
          const float* __restrict__ bb, float* __restrict__ out) {
  int o = blockIdx.x * blockDim.x + threadIdx.x;
  if (o >= BATCH * 2 * EMBED) return;
  int b = o / (2 * EMBED), c = o % (2 * EMBED);
  float a = bb[c];
  for (int k = 0; k < EMBED; ++k) {
    float t = temb[b * EMBED + k];
    a += (t / (1.f + expf(-t))) * w[(long)c * EMBED + k];
  }
  out[o] = a;
}

// ---------------- CPB bias ----------------
__global__ __launch_bounds__(256)
void k_cpb(const float* __restrict__ w1, const float* __restrict__ b1,
           const float* __restrict__ w2, float* __restrict__ out /*[289,24]*/) {
  int e = blockIdx.x;                     // 0..288
  int dh = e / 17 - 8, dw = e % 17 - 8;
  float inv = 1.0f / log2f(8.0f);
  float ax = fabsf((float)dh), ay = fabsf((float)dw);
  float cx = ((dh > 0) - (dh < 0)) * log2f(ax + 1.f) * inv;
  float cy = ((dw > 0) - (dw < 0)) * log2f(ay + 1.f) * inv;
  __shared__ float hbuf[512];
  int tid = threadIdx.x;
  for (int j = tid; j < 512; j += 256)
    hbuf[j] = fmaxf(0.f, w1[j * 2] * cx + w1[j * 2 + 1] * cy + b1[j]);
  __syncthreads();
  if (tid < HEADS) {
    float a = 0.f;
    for (int j = 0; j < 512; ++j) a += w2[tid * 512 + j] * hbuf[j];
    out[e * HEADS + tid] = a;
  }
}

__global__ __launch_bounds__(256)
void k_bias16(const float* __restrict__ cpb, float* __restrict__ b16 /*[24,81,81]*/) {
  long n = (long)HEADS * NTOK * NTOK;
  long i = (long)blockIdx.x * blockDim.x + threadIdx.x;
  if (i >= n) return;
  int h = (int)(i / (NTOK * NTOK)); int r = (int)(i % (NTOK * NTOK));
  int ti = r / NTOK, tm = r % NTOK;
  int ri = ti / WS9, ci = ti % WS9, rm = tm / WS9, cm = tm % WS9;
  int idx = (ri - rm + 8) * 17 + (ci - cm + 8);
  float v = cpb[idx * HEADS + h];
  b16[i] = 16.0f / (1.0f + expf(-v));
}

// ---------------- attention core (cosine attn, per window/head) ----------------
__global__ __launch_bounds__(128)
void k_attn(const float* __restrict__ qkv, const float* __restrict__ b16,
            const float* __restrict__ ls, _Float16* __restrict__ out, int shifted) {
  __shared__ float kn[NTOK * HEAD_DIM];   // normalized K
  __shared__ float pr[NTOK * NTOK];       // logits / probs (row per token)
  int win = blockIdx.x, head = blockIdx.y;
  int tid = threadIdx.x;
  long base = (long)win * NTOK * (3 * EMBED);
  float scale = expf(fminf(ls[head], LOGIT_MAX));
  int wr = win % 18, wh = wr / 6, ww = wr % 6;

  float q[HEAD_DIM];
  if (tid < NTOK) {
    const float* qp = qkv + base + (long)tid * (3 * EMBED) + head * HEAD_DIM;
    const float* kp = qp + EMBED;
    float qs = 1e-12f, ks = 1e-12f;
#pragma unroll
    for (int d = 0; d < HEAD_DIM; ++d) { q[d] = qp[d]; qs += q[d] * q[d]; }
#pragma unroll
    for (int d = 0; d < HEAD_DIM; ++d) { float kv = kp[d]; ks += kv * kv; }
    float krs = rsqrtf(ks), qrs = rsqrtf(qs);
#pragma unroll
    for (int d = 0; d < HEAD_DIM; ++d) { kn[tid * HEAD_DIM + d] = kp[d] * krs; q[d] *= qrs; }
  }
  __syncthreads();
  if (tid < NTOK) {
    int hi = wh * WS9 + tid / WS9, wi = ww * WS9 + tid % WS9;
    int ridi = (hi < 18 ? 0 : (hi < 23 ? 1 : 2)) * 3 + (wi < 45 ? 0 : (wi < 50 ? 1 : 2));
    float mx = -1e30f;
    for (int m = 0; m < NTOK; ++m) {
      float dot = 0.f;
#pragma unroll
      for (int d = 0; d < HEAD_DIM; ++d) dot += q[d] * kn[m * HEAD_DIM + d];
      float l = dot * scale + b16[((long)head * NTOK + tid) * NTOK + m];
      if (shifted) {
        int hm = wh * WS9 + m / WS9, wm = ww * WS9 + m % WS9;
        int ridm = (hm < 18 ? 0 : (hm < 23 ? 1 : 2)) * 3 + (wm < 45 ? 0 : (wm < 50 ? 1 : 2));
        if (ridm != ridi) l -= 100.0f;
      }
      pr[tid * NTOK + m] = l;
      mx = fmaxf(mx, l);
    }
    float sum = 0.f;
    for (int m = 0; m < NTOK; ++m) {
      float e = expf(pr[tid * NTOK + m] - mx);
      pr[tid * NTOK + m] = e; sum += e;
    }
    float invs = 1.0f / sum;
    const float* vbase = qkv + base + 2 * EMBED + head * HEAD_DIM;
#pragma unroll 4
    for (int d = 0; d < HEAD_DIM; ++d) {
      float a = 0.f;
      for (int m = 0; m < NTOK; ++m)
        a += pr[tid * NTOK + m] * vbase[(long)m * (3 * EMBED) + d];
      out[((long)win * NTOK + tid) * EMBED + head * HEAD_DIM + d] = (_Float16)(a * invs);
    }
  }
}

// ---------------- weight repacks ----------------
__global__ __launch_bounds__(256)
void k_repack_33(const float* __restrict__ s, _Float16* __restrict__ d, int Co, int Ci) {
  long n = (long)Co * Ci * 9;
  long i = (long)blockIdx.x * blockDim.x + threadIdx.x;
  if (i >= n) return;
  int tap = (int)(i % 9); long r = i / 9;
  int ci = (int)(r % Ci); int co = (int)(r / Ci);
  d[((long)tap * Co + co) * Ci + ci] = (_Float16)s[i];     // s: [co][ci][tap]
}

__global__ __launch_bounds__(256)
void k_repack_up(const float* __restrict__ s, _Float16* __restrict__ d) {
  long n = (long)EMBED * EMBED * 16;
  long i = (long)blockIdx.x * blockDim.x + threadIdx.x;
  if (i >= n) return;
  int tap = (int)(i & 15); long r = i >> 4;
  int co = (int)(r % EMBED); int ci = (int)(r / EMBED);    // s: [ci][co][tap]
  d[((long)tap * EMBED + co) * EMBED + ci] = (_Float16)s[i];
}

// ======================= host orchestration =======================
struct Blk {
  const float *qkv_w, *q_bias, *v_bias, *ls, *cpb_w1, *cpb_b1, *cpb_w2,
              *proj_w, *proj_b, *n1w, *n1b, *fc1w, *fc1b, *fc2w, *fc2b, *n2w, *n2b;
};

static inline unsigned gblk(long n) { return (unsigned)((n + 255) / 256); }

static void cvt16(const float* s, _Float16* d, long n, hipStream_t st) {
  k_f32_to_f16<<<gblk(n), 256, 0, st>>>(s, d, n);
}

static void gemm(hipStream_t st, const _Float16* A, int lda, const _Float16* B,
                 long ldbn, const float* bias, float* C, int ldc,
                 int M, int N, int K, int acc, int act) {
  dim3 g((M + BM - 1) / BM, (N + BN - 1) / BN);
  k_gemm_nt<<<g, 256, 0, st>>>(A, lda, B, ldbn, bias, C, ldc, M, N, K, acc, act);
}

extern "C" void kernel_launch(void* const* d_in, const int* in_sizes, int n_in,
                              void* d_out, int out_size, void* d_ws, size_t ws_size,
                              hipStream_t stream) {
  (void)in_sizes; (void)n_in; (void)ws_size;
  const float* z_high = (const float*)d_in[0];
  const float* temb   = (const float*)d_in[1];
  const float* skip_in= (const float*)d_in[2];
  const float* zexp_w = (const float*)d_in[3];
  const float* zexp_b = (const float*)d_in[4];
  Blk blk[4];
  for (int i = 0; i < 4; ++i) {
    int o = 5 + i * 17;
    blk[i] = Blk{ (const float*)d_in[o+0], (const float*)d_in[o+1], (const float*)d_in[o+2],
                  (const float*)d_in[o+3], (const float*)d_in[o+4], (const float*)d_in[o+5],
                  (const float*)d_in[o+6], (const float*)d_in[o+7], (const float*)d_in[o+8],
                  (const float*)d_in[o+9], (const float*)d_in[o+10],(const float*)d_in[o+11],
                  (const float*)d_in[o+12],(const float*)d_in[o+13],(const float*)d_in[o+14],
                  (const float*)d_in[o+15],(const float*)d_in[o+16] };
  }
  const float* fusion_w = (const float*)d_in[73];
  const float* fusion_b = (const float*)d_in[74];
  const float* n1_w = (const float*)d_in[75]; const float* n1_b = (const float*)d_in[76];
  const float* c1_w = (const float*)d_in[77]; const float* c1_b = (const float*)d_in[78];
  const float* emb_w= (const float*)d_in[79]; const float* emb_b= (const float*)d_in[80];
  const float* n2_w = (const float*)d_in[81]; const float* n2_b = (const float*)d_in[82];
  const float* c2_w = (const float*)d_in[83]; const float* c2_b = (const float*)d_in[84];
  const float* skw  = (const float*)d_in[85]; const float* skb  = (const float*)d_in[86];
  const float* up_w = (const float*)d_in[87]; const float* up_b = (const float*)d_in[88];
  const float* head_w = (const float*)d_in[89]; const float* head_b = (const float*)d_in[90];

  // ---- workspace carve ----
  char* wsb = (char*)d_ws;
  size_t off = 0;
  auto carve = [&](size_t bytes) -> void* {
    void* p = wsb + off;
    off += (bytes + 255) & ~(size_t)255;
    return p;
  };
  _Float16* W16   = (_Float16*)carve((size_t)9 * EMBED * (2*EMBED) * 2);   // 85 MB (max repack)
  _Float16* A16   = (_Float16*)carve((size_t)TUP * EMBED * 2);             // 36 MB
  float*    BIG   = (float*)carve((size_t)TUP * 2240 * 4);                 // 105 MB
  float*    x_cur = (float*)carve((size_t)T_TOK * EMBED * 4);
  float*    s2b   = (float*)carve((size_t)T_TOK * EMBED * 4);
  float*    projm = (float*)carve((size_t)T_TOK * EMBED * 4);
  float*    tokA  = (float*)carve((size_t)T_TOK * EMBED * 4);
  float*    tokSk = (float*)carve((size_t)T_TOK * EMBED * 4);
  float*    tokC  = (float*)carve((size_t)T_TOK * EMBED * 4);
  _Float16* xw16  = (_Float16*)carve((size_t)T_TOK * EMBED * 2);
  _Float16* at16  = (_Float16*)carve((size_t)T_TOK * EMBED * 2);
  _Float16* x16   = (_Float16*)carve((size_t)T_TOK * EMBED * 2);
  _Float16* glu16 = (_Float16*)carve((size_t)T_TOK * MLP_HID * 2);
  _Float16* cat16 = (_Float16*)carve((size_t)T_TOK * 2 * EMBED * 2);
  float*    xcat  = (float*)carve((size_t)BATCH * 2 * EMBED * H_LAT * W_LAT * 4);
  float*    gno   = (float*)carve((size_t)BATCH * 2 * EMBED * H_LAT * W_LAT * 4);
  float*    c1n   = (float*)carve((size_t)BATCH * EMBED * H_LAT * W_LAT * 4);
  float*    resn  = (float*)carve((size_t)BATCH * EMBED * H_LAT * W_LAT * 4);
  float*    hup   = (float*)carve((size_t)BATCH * EMBED * 54 * 108 * 4);
  float*    qkvb  = (float*)carve(4608 * 4);
  float*    cpbb  = (float*)carve(289 * HEADS * 4);
  float*    b16b  = (float*)carve((size_t)HEADS * NTOK * NTOK * 4);
  float*    ssb   = (float*)carve(BATCH * 2 * EMBED * 4);

  const int T = T_TOK;

  // ---- z expansion: conv1x1 128->1536 ----
  k_gather_tap<<<gblk((long)T * 128), 256, 0, stream>>>(z_high, A16, BATCH, 128, H_LAT, W_LAT, 0, 0);
  cvt16(zexp_w, W16, (long)EMBED * 128, stream);
  gemm(stream, A16, 128, W16, 128, zexp_b, x_cur, EMBED, T, EMBED, 128, 0, 0);

  // ---- 4 swin blocks ----
  for (int bi = 0; bi < 4; ++bi) {
    const Blk& p = blk[bi];
    int shift = (bi & 1) ? SHIFT4 : 0;
    k_gather_win<<<gblk((long)T * EMBED), 256, 0, stream>>>(x_cur, xw16, shift);
    cvt16(p.qkv_w, W16, (long)3 * EMBED * EMBED, stream);
    k_pack_qkvb<<<gblk(4608), 256, 0, stream>>>(p.q_bias, p.v_bias, qkvb);
    gemm(stream, xw16, EMBED, W16, EMBED, qkvb, BIG, 3 * EMBED, T, 3 * EMBED, EMBED, 0, 0);
    k_cpb<<<289, 256, 0, stream>>>(p.cpb_w1, p.cpb_b1, p.cpb_w2, cpbb);
    k_bias16<<<gblk((long)HEADS * NTOK * NTOK), 256, 0, stream>>>(cpbb, b16b);
    k_attn<<<dim3(36, HEADS), 128, 0, stream>>>(BIG, b16b, p.ls, at16, shift ? 1 : 0);
    cvt16(p.proj_w, W16, (long)EMBED * EMBED, stream);
    gemm(stream, at16, EMBED, W16, EMBED, p.proj_b, projm, EMBED, T, EMBED, EMBED, 0, 0);
    k_add_ln<<<T, 256, 0, stream>>>(projm, x_cur, x_cur, p.n1w, p.n1b, 1, shift);
    cvt16(x_cur, x16, (long)T * EMBED, stream);
    cvt16(p.fc1w, W16, (long)2 * MLP_HID * EMBED, stream);
    gemm(stream, x16, EMBED, W16, EMBED, p.fc1b, BIG, 2 * MLP_HID, T, 2 * MLP_HID, EMBED, 0, 0);
    k_glu<<<gblk((long)T * MLP_HID), 256, 0, stream>>>(BIG, glu16);
    cvt16(p.fc2w, W16, (long)EMBED * MLP_HID, stream);
    gemm(stream, glu16, MLP_HID, W16, MLP_HID, p.fc2b, projm, EMBED, T, EMBED, MLP_HID, 0, 0);
    k_add_ln<<<T, 256, 0, stream>>>(projm, x_cur, x_cur, p.n2w, p.n2b, 0, 0);
    if (bi == 1) k_copy_f32<<<gblk((long)T * EMBED), 256, 0, stream>>>(x_cur, s2b, (long)T * EMBED);
  }

  // ---- fusion ----
  k_concat_tok_f16<<<gblk((long)T * 2 * EMBED), 256, 0, stream>>>(s2b, x_cur, cat16);
  cvt16(fusion_w, W16, (long)EMBED * 2 * EMBED, stream);
  gemm(stream, cat16, 2 * EMBED, W16, 2 * EMBED, fusion_b, tokA, EMBED, T, EMBED, 2 * EMBED, 0, 0);
  k_concat_nchw<<<gblk((long)BATCH * 2 * EMBED * H_LAT * W_LAT), 256, 0, stream>>>(tokA, skip_in, xcat);

  // ---- resblock ----
  k_gather_tap<<<gblk((long)T * 2 * EMBED), 256, 0, stream>>>(xcat, A16, BATCH, 2 * EMBED, H_LAT, W_LAT, 0, 0);
  cvt16(skw, W16, (long)EMBED * 2 * EMBED, stream);
  gemm(stream, A16, 2 * EMBED, W16, 2 * EMBED, skb, tokSk, EMBED, T, EMBED, 2 * EMBED, 0, 0);
  k_gn<<<BATCH * 32, 256, 0, stream>>>(xcat, n1_w, n1_b, nullptr, gno, 2 * EMBED, 1);
  k_repack_33<<<gblk((long)EMBED * 2 * EMBED * 9), 256, 0, stream>>>(c1_w, W16, EMBED, 2 * EMBED);
  for (int tap = 0; tap < 9; ++tap) {
    int dy = tap / 3 - 1, dx = tap % 3 - 1;
    k_gather_tap<<<gblk((long)T * 2 * EMBED), 256, 0, stream>>>(gno, A16, BATCH, 2 * EMBED, H_LAT, W_LAT, dy, dx);
    gemm(stream, A16, 2 * EMBED, W16 + (size_t)tap * EMBED * 2 * EMBED, 2 * EMBED,
         (tap == 8) ? c1_b : nullptr, tokC, EMBED, T, EMBED, 2 * EMBED, tap > 0, 0);
  }
  k_tok_to_nchw<<<gblk((long)BATCH * EMBED * H_LAT * W_LAT), 256, 0, stream>>>(tokC, c1n, nullptr, EMBED);
  k_ss<<<gblk(BATCH * 2 * EMBED), 256, 0, stream>>>(temb, emb_w, emb_b, ssb);
  k_gn<<<BATCH * 32, 256, 0, stream>>>(c1n, n2_w, n2_b, ssb, gno, EMBED, 1);
  k_repack_33<<<gblk((long)EMBED * EMBED * 9), 256, 0, stream>>>(c2_w, W16, EMBED, EMBED);
  for (int tap = 0; tap < 9; ++tap) {
    int dy = tap / 3 - 1, dx = tap % 3 - 1;
    k_gather_tap<<<gblk((long)T * EMBED), 256, 0, stream>>>(gno, A16, BATCH, EMBED, H_LAT, W_LAT, dy, dx);
    gemm(stream, A16, EMBED, W16 + (size_t)tap * EMBED * EMBED, EMBED,
         (tap == 8) ? c2_b : nullptr, tokC, EMBED, T, EMBED, EMBED, tap > 0, 0);
  }
  k_tok_to_nchw<<<gblk((long)BATCH * EMBED * H_LAT * W_LAT), 256, 0, stream>>>(tokC, resn, tokSk, EMBED);

  // ---- transposed conv 4x4 s2 p1 ----
  k_repack_up<<<gblk((long)EMBED * EMBED * 16), 256, 0, stream>>>(up_w, W16);
  for (int pa = 0; pa < 2; ++pa) {
    for (int pb = 0; pb < 2; ++pb) {
      int first = 1;
      int khs[2]; khs[0] = pa ? 0 : 1; khs[1] = pa ? 2 : 3;
      int kws[2]; kws[0] = pb ? 0 : 1; kws[1] = pb ? 2 : 3;
      for (int a = 0; a < 2; ++a) {
        for (int c = 0; c < 2; ++c) {
          int kh = khs[a], kw = kws[c];
          int dy = (pa + 1 - kh) / 2, dx = (pb + 1 - kw) / 2;
          k_gather_tap<<<gblk((long)T * EMBED), 256, 0, stream>>>(resn, A16, BATCH, EMBED, H_LAT, W_LAT, dy, dx);
          gemm(stream, A16, EMBED, W16 + (size_t)(kh * 4 + kw) * EMBED * EMBED, EMBED,
               nullptr, tokC, EMBED, T, EMBED, EMBED, first ? 0 : 1, 0);
          first = 0;
        }
      }
      k_scatterT<<<gblk((long)T * EMBED), 256, 0, stream>>>(tokC, hup, pa, pb, up_b);
    }
  }

  // ---- head ----
  k_gather_tap<<<gblk((long)TUP * EMBED), 256, 0, stream>>>(hup, A16, BATCH, EMBED, 54, 108, 0, 0);
  cvt16(head_w, W16, (long)2240 * EMBED, stream);
  gemm(stream, A16, EMBED, W16, EMBED, head_b, BIG, 2240, TUP, 2240, EMBED, 0, 0);
  k_head_reorder<<<gblk((long)out_size), 256, 0, stream>>>(BIG, (float*)d_out, (long)out_size);
}
